// LinearAttention_51634096833044
// MI455X (gfx1250) — matrix operand order repaired and checked by reading, not compiled, not run on hardware
//
#include <hip/hip_runtime.h>
#include <cstddef>
#include <cstdint>

typedef __bf16 bf16_t;
typedef __attribute__((ext_vector_type(16))) __bf16 v16bf;
typedef __attribute__((ext_vector_type(8)))  __bf16 v8bf;
typedef __attribute__((ext_vector_type(8)))  float  v8f;

typedef unsigned int u32x4 __attribute__((ext_vector_type(4)));
typedef int          i32x4 __attribute__((ext_vector_type(4)));
typedef int          i32x8 __attribute__((ext_vector_type(8)));

#define EPS_F 1e-8f

#if defined(__HIP_DEVICE_COMPILE__) && __has_builtin(__builtin_amdgcn_tensor_load_to_lds) && \
    __has_builtin(__builtin_amdgcn_s_wait_tensorcnt)
#define USE_TDM 1
#else
#define USE_TDM 0
#endif

// ---------------------------------------------------------------------------
// WMMA fragment helpers (wave32, 16x16x32 bf16, f32 accumulate)
// A (16x32 bf16): lane L<16 -> row M=L, K = kb+0..15
//                 lane L>=16 -> row M=L-16, K = kb+16..31
//   element e:    K = kb + e + 16*hi
// B (32x16 bf16): lane L -> col N=L&15, K = kb + e + 16*hi
// C/D (16x16 f32): elem r -> M = r + 8*hi, N = lane&15
// ---------------------------------------------------------------------------

__device__ __forceinline__ int koff_a(int e, int hi) {
    return e + hi * 16;
}

// A fragment from row-major bf16 (two contiguous 8-elem runs -> 2 x b128)
__device__ __forceinline__ v16bf load_a_frag(const bf16_t* src, int ld, int row, int kb) {
    const int lane = threadIdx.x & 31;
    const int hi   = (lane >> 4) & 1;
    const int m    = row + (lane & 15);
    const bf16_t* p = src + (size_t)m * ld + kb + hi * 16;
    v8bf lo  = *reinterpret_cast<const v8bf*>(p);
    v8bf hi8 = *reinterpret_cast<const v8bf*>(p + 8);
    v16bf a;
#pragma unroll
    for (int e = 0; e < 8; ++e) { a[e] = lo[e]; a[e + 8] = hi8[e]; }
    return a;
}

// B fragment where logical B[k][n] = src[n*ld + k] (src N-major, K contiguous).
// One contiguous 16-elem run per lane.
__device__ __forceinline__ v16bf load_bT_frag(const bf16_t* src, int ld, int col, int kb) {
    const int lane = threadIdx.x & 31;
    const int hi   = (lane >> 4) & 1;
    const int n    = col + (lane & 15);
    return *reinterpret_cast<const v16bf*>(src + (size_t)n * ld + kb + hi * 16);
}

// A fragment of a transposed source: A[m][k] = src[k*ld + m]  (strided, scalar)
__device__ __forceinline__ v16bf lds_aT_frag(const bf16_t* src, int ld, int row, int kb) {
    const int lane = threadIdx.x & 31;
    const int hi   = (lane >> 4) & 1;
    const int m    = row + (lane & 15);
    v16bf a;
#pragma unroll
    for (int e = 0; e < 16; ++e) a[e] = src[(size_t)(kb + koff_a(e, hi)) * ld + m];
    return a;
}

// B fragment, row-major source: B[k][n] = src[k*ld + n]  (strided, scalar)
__device__ __forceinline__ v16bf lds_b_frag(const bf16_t* src, int ld, int kb, int col) {
    const int lane = threadIdx.x & 31;
    const int hi   = (lane >> 4) & 1;
    const int n    = col + (lane & 15);
    v16bf b;
#pragma unroll
    for (int e = 0; e < 16; ++e) b[e] = src[(size_t)(kb + e + hi * 16) * ld + n];
    return b;
}

// B fragment from f32 source (kv state), converted to bf16 on read
__device__ __forceinline__ v16bf lds_b_frag_f32(const float* src, int ld, int kb, int col) {
    const int lane = threadIdx.x & 31;
    const int hi   = (lane >> 4) & 1;
    const int n    = col + (lane & 15);
    v16bf b;
#pragma unroll
    for (int e = 0; e < 16; ++e) b[e] = (bf16_t)src[(size_t)(kb + e + hi * 16) * ld + n];
    return b;
}

__device__ __forceinline__ v8f lds_c_load_f32(const float* src, int ld, int row, int col) {
    const int lane = threadIdx.x & 31;
    const int hi   = (lane >> 4) & 1;
    const int n    = col + (lane & 15);
    v8f c;
#pragma unroll
    for (int r = 0; r < 8; ++r) c[r] = src[(size_t)(row + r + hi * 8) * ld + n];
    return c;
}

__device__ __forceinline__ void lds_c_store_f32(float* dst, int ld, int row, int col, v8f c) {
    const int lane = threadIdx.x & 31;
    const int hi   = (lane >> 4) & 1;
    const int n    = col + (lane & 15);
#pragma unroll
    for (int r = 0; r < 8; ++r) dst[(size_t)(row + r + hi * 8) * ld + n] = c[r];
}

// ---------------------------------------------------------------------------
// Tensor Data Mover: async-load one [rows x 64] bf16 tile (row stride 1024
// elements) from global into LDS.  Descriptor packing per
// CDNA5 ISA sect. 8.3/8.4 (2-D tensor; groups 2/3 zero;
// clang-23 6-arg builtin form).
// ---------------------------------------------------------------------------
#if USE_TDM
__device__ __forceinline__ void tdm_load_tile_bf16(const bf16_t* gsrc,
                                                   const bf16_t* lds_dst,
                                                   int rows) {
    const unsigned long long ga = (unsigned long long)(uintptr_t)gsrc;
    const unsigned lds_off = (unsigned)(uintptr_t)lds_dst;  // low 32 bits = LDS byte addr
    u32x4 g0;
    g0[0] = 1u;                                  // count=1 valid descriptor
    g0[1] = lds_off;                             // lds_addr
    g0[2] = (unsigned)(ga & 0xFFFFFFFFu);        // global_addr[31:0]
    g0[3] = (unsigned)((ga >> 32) & 0x1FFFFFFu)  // global_addr[56:32]
            | (2u << 30);                        // type = 2 (image)
    i32x8 g1;
    g1[0] = (1 << 16);                           // data_size=1 (2 bytes/elem)
    g1[1] = (64 << 16);                          // tensor_dim0 = 64 (lo16 of field)
    g1[2] = (4096 << 16);                        // tensor_dim0 hi | tensor_dim1 lo
    g1[3] = (64 << 16);                          // tensor_dim1 hi | tile_dim0 = 64
    g1[4] = rows & 0xFFFF;                       // tile_dim1 = rows, tile_dim2 = 0
    g1[5] = 1024;                                // tensor_dim0_stride[31:0] = 1024
    g1[6] = 0;                                   // stride0 hi | stride1 lo
    g1[7] = 0;
    const i32x4 z4 = {0, 0, 0, 0};
    const i32x8 z8 = {0, 0, 0, 0, 0, 0, 0, 0};
    __builtin_amdgcn_tensor_load_to_lds(g0, g1, z4, z4, z8, 0);
}
#endif

// ---------------------------------------------------------------------------
// f32 -> bf16 conversion
// ---------------------------------------------------------------------------
__global__ void f32_to_bf16_kernel(const float* __restrict__ in,
                                   bf16_t* __restrict__ out, size_t n) {
    size_t i      = (size_t)blockIdx.x * blockDim.x + threadIdx.x;
    size_t stride = (size_t)gridDim.x * blockDim.x;
    for (; i < n; i += stride) out[i] = (bf16_t)in[i];
}

// ---------------------------------------------------------------------------
// GEMM: Y[M,N] = X[M,K] @ W[N,K]^T   (bf16 in, f32 WMMA accumulate)
// mode 0: store bf16 | mode 1: elu(y)+1 -> bf16 | mode 2: store f32
// grid: (M/256, N/64), block: 256 threads (8 waves), wave tile 32x64
// ---------------------------------------------------------------------------
__global__ __launch_bounds__(256) void gemm_bf16_kernel(
    const bf16_t* __restrict__ X, const bf16_t* __restrict__ W,
    void* __restrict__ Yv, int K, int N, int mode) {
    const int lane    = threadIdx.x & 31;
    const int wave    = threadIdx.x >> 5;
    const int rowBase = blockIdx.x * 256 + wave * 32;
    const int colBase = blockIdx.y * 64;

    v8f acc[2][4];
#pragma unroll
    for (int mt = 0; mt < 2; ++mt)
#pragma unroll
        for (int nt = 0; nt < 4; ++nt) acc[mt][nt] = (v8f){0,0,0,0,0,0,0,0};

    for (int kb = 0; kb < K; kb += 32) {
        v16bf a0 = load_a_frag(X, K, rowBase,      kb);
        v16bf a1 = load_a_frag(X, K, rowBase + 16, kb);
#pragma unroll
        for (int nt = 0; nt < 4; ++nt) {
            v16bf b = load_bT_frag(W, K, colBase + nt * 16, kb);
            acc[0][nt] = __builtin_amdgcn_wmma_f32_16x16x32_bf16(
                false, a0, false, b, (short)0, acc[0][nt], false, false);
            acc[1][nt] = __builtin_amdgcn_wmma_f32_16x16x32_bf16(
                false, a1, false, b, (short)0, acc[1][nt], false, false);
        }
    }

    const int hi  = (lane >> 4) & 1;
    const int nlo = lane & 15;
#pragma unroll
    for (int mt = 0; mt < 2; ++mt)
#pragma unroll
        for (int nt = 0; nt < 4; ++nt)
#pragma unroll
            for (int r = 0; r < 8; ++r) {
                const int m = rowBase + mt * 16 + r + hi * 8;
                const int n = colBase + nt * 16 + nlo;
                float y = acc[mt][nt][r];
                if (mode == 1) y = (y > 0.0f) ? (y + 1.0f) : __expf(y);  // elu(y)+1
                if (mode == 2) ((float*)Yv)[(size_t)m * N + n] = y;
                else           ((bf16_t*)Yv)[(size_t)m * N + n] = (bf16_t)y;
            }
}

// ---------------------------------------------------------------------------
// Chunked linear-attention scan.  One wave32 per (b,h); chunk C=32, Dh=64.
// q/k/v chunk tiles double-buffered in LDS via TDM async loads; kv_state
// [64x64] f32 + ksum [64] persist in LDS across the 128 chunk iterations.
//   S      = tril(q_c @ k_cT)            (2x2 tiles, K=64)
//   out    = (q_c @ kv + S @ v_c) / den  (2x4 tiles)
//   den_i  = sum_j S[i][j] + q_i . ksum + eps
//   kv    += k_cT @ v_c                  (4x4 tiles, K=32)
// ---------------------------------------------------------------------------
__global__ __launch_bounds__(32) void linattn_scan_kernel(
    const bf16_t* __restrict__ Q, const bf16_t* __restrict__ Kc,
    const bf16_t* __restrict__ V, bf16_t* __restrict__ O,
    int S, int D) {
    const int bh   = blockIdx.x;          // 0..31
    const int b    = bh >> 4;
    const int h    = bh & 15;
    const int lane = threadIdx.x;         // blockDim.x == 32
    const int hi   = (lane >> 4) & 1;
    const int nlo  = lane & 15;

    __shared__ float  kv[64 * 68];        // [64][68] padded f32 state
    __shared__ float  ksum[64];
    __shared__ bf16_t qs[2][32 * 64];     // double-buffered chunk tiles
    __shared__ bf16_t ks[2][32 * 64];
    __shared__ bf16_t vs[2][32 * 64];
    __shared__ bf16_t Ss[32 * 40];        // masked S, ld=40 keeps 16B alignment
    __shared__ float  den[32];

    for (int d = lane; d < 64; d += 32) {
        ksum[d] = 0.0f;
#pragma unroll
        for (int e = 0; e < 68; ++e) kv[d * 68 + e] = 0.0f;
    }

    const size_t base = (size_t)b * S * D + (size_t)h * 64;
    const int    NC   = S / 32;

#if USE_TDM
    {   // prefetch chunk 0 into buffer 0
        const size_t g0 = base;
        tdm_load_tile_bf16(Q  + g0, &qs[0][0], 32);
        tdm_load_tile_bf16(Kc + g0, &ks[0][0], 32);
        tdm_load_tile_bf16(V  + g0, &vs[0][0], 32);
    }
#endif

    for (int c = 0; c < NC; ++c) {
        const int buf = c & 1;
#if USE_TDM
        if (c + 1 < NC) {   // async-prefetch next chunk, then wait for current
            const size_t gn = base + (size_t)((c + 1) * 32) * D;
            const int nb = (c + 1) & 1;
            tdm_load_tile_bf16(Q  + gn, &qs[nb][0], 32);
            tdm_load_tile_bf16(Kc + gn, &ks[nb][0], 32);
            tdm_load_tile_bf16(V  + gn, &vs[nb][0], 32);
            __builtin_amdgcn_s_wait_tensorcnt(3);
        } else {
            __builtin_amdgcn_s_wait_tensorcnt(0);
        }
#else
        {   // fallback: one row per lane, 4 x 32B loads per tensor
            const size_t rb = base + (size_t)(c * 32 + lane) * D;
#pragma unroll
            for (int t = 0; t < 4; ++t) {
                *reinterpret_cast<v16bf*>(&qs[buf][lane * 64 + t * 16]) =
                    reinterpret_cast<const v16bf*>(Q + rb)[t];
                *reinterpret_cast<v16bf*>(&ks[buf][lane * 64 + t * 16]) =
                    reinterpret_cast<const v16bf*>(Kc + rb)[t];
                *reinterpret_cast<v16bf*>(&vs[buf][lane * 64 + t * 16]) =
                    reinterpret_cast<const v16bf*>(V + rb)[t];
            }
        }
#endif
        const bf16_t* qc = &qs[buf][0];
        const bf16_t* kc = &ks[buf][0];
        const bf16_t* vc = &vs[buf][0];

        // --- S = tril(q @ kT) -> bf16 in LDS ---
#pragma unroll
        for (int it = 0; it < 2; ++it)
#pragma unroll
            for (int jt = 0; jt < 2; ++jt) {
                v8f s = (v8f){0,0,0,0,0,0,0,0};
#pragma unroll
                for (int kt = 0; kt < 2; ++kt) {
                    v16bf a  = load_a_frag(qc, 64, it * 16, kt * 32);
                    v16bf bT = load_bT_frag(kc, 64, jt * 16, kt * 32); // B[d][j]=k[j][d]
                    s = __builtin_amdgcn_wmma_f32_16x16x32_bf16(
                        false, a, false, bT, (short)0, s, false, false);
                }
#pragma unroll
                for (int r = 0; r < 8; ++r) {
                    const int i = it * 16 + r + hi * 8;
                    const int j = jt * 16 + nlo;
                    Ss[i * 40 + j] = (bf16_t)((j <= i) ? s[r] : 0.0f);  // causal incl. diag
                }
            }

        // --- denominator per row ---
        {
            float dsum = EPS_F;
#pragma unroll
            for (int j = 0; j < 32; ++j) dsum += (float)Ss[lane * 40 + j];
#pragma unroll
            for (int d = 0; d < 64; ++d) dsum += (float)qc[lane * 64 + d] * ksum[d];
            den[lane] = dsum;
        }

        // --- out = (q @ kv + S @ v) / den ---
#pragma unroll
        for (int mt = 0; mt < 2; ++mt)
#pragma unroll
            for (int nt = 0; nt < 4; ++nt) {
                v8f o = (v8f){0,0,0,0,0,0,0,0};
                {   // intra-chunk: S @ v   (K=32)
                    v16bf a  = load_a_frag(Ss, 40, mt * 16, 0);
                    v16bf bv = lds_b_frag(vc, 64, 0, nt * 16);
                    o = __builtin_amdgcn_wmma_f32_16x16x32_bf16(
                        false, a, false, bv, (short)0, o, false, false);
                }
#pragma unroll
                for (int kt = 0; kt < 2; ++kt) {  // inter-chunk: q @ kv_state
                    v16bf aq  = load_a_frag(qc, 64, mt * 16, kt * 32);
                    v16bf bkv = lds_b_frag_f32(kv, 68, kt * 32, nt * 16);
                    o = __builtin_amdgcn_wmma_f32_16x16x32_bf16(
                        false, aq, false, bkv, (short)0, o, false, false);
                }
#pragma unroll
                for (int r = 0; r < 8; ++r) {
                    const int i = mt * 16 + r + hi * 8;
                    const int e = nt * 16 + nlo;
                    const float val = o[r] / den[i];
                    O[base + (size_t)(c * 32 + i) * D + e] = (bf16_t)val;
                }
            }

        // --- kv_state += kT @ v   (M=64, N=64, K=32) ---
#pragma unroll
        for (int dt = 0; dt < 4; ++dt)
#pragma unroll
            for (int et = 0; et < 4; ++et) {
                v8f cf = lds_c_load_f32(kv, 68, dt * 16, et * 16);
                v16bf ak = lds_aT_frag(kc, 64, dt * 16, 0);   // A[d][kk]=k[kk][d]
                v16bf bv = lds_b_frag(vc, 64, 0, et * 16);
                cf = __builtin_amdgcn_wmma_f32_16x16x32_bf16(
                    false, ak, false, bv, (short)0, cf, false, false);
                lds_c_store_f32(kv, 68, dt * 16, et * 16, cf);
            }

        // --- ksum += column sums of k chunk ---
        for (int d = lane; d < 64; d += 32) {
            float s2 = ksum[d];
#pragma unroll
            for (int i = 0; i < 32; ++i) s2 += (float)kc[i * 64 + d];
            ksum[d] = s2;
        }
    }
}

// ---------------------------------------------------------------------------
// Host launcher
// ---------------------------------------------------------------------------
extern "C" void kernel_launch(void* const* d_in, const int* in_sizes, int n_in,
                              void* d_out, int out_size, void* d_ws, size_t ws_size,
                              hipStream_t stream) {
    (void)in_sizes; (void)n_in; (void)out_size; (void)ws_size;
    const float* x  = (const float*)d_in[0];
    const float* Wq = (const float*)d_in[1];
    const float* Wk = (const float*)d_in[2];
    const float* Wv = (const float*)d_in[3];
    const float* Wo = (const float*)d_in[4];

    const int B = 2, S = 4096, D = 1024;
    const size_t M  = (size_t)B * S;   // 8192
    const size_t xN = M * D;           // 8,388,608
    const size_t wN = (size_t)D * D;   // 1,048,576

    char* ws = (char*)d_ws;
    bf16_t* xb  = (bf16_t*)ws; ws += xN * sizeof(bf16_t);
    bf16_t* wqb = (bf16_t*)ws; ws += wN * sizeof(bf16_t);
    bf16_t* wkb = (bf16_t*)ws; ws += wN * sizeof(bf16_t);
    bf16_t* wvb = (bf16_t*)ws; ws += wN * sizeof(bf16_t);
    bf16_t* wob = (bf16_t*)ws; ws += wN * sizeof(bf16_t);
    bf16_t* qb  = (bf16_t*)ws; ws += xN * sizeof(bf16_t);
    bf16_t* kb  = (bf16_t*)ws; ws += xN * sizeof(bf16_t);
    bf16_t* vb  = (bf16_t*)ws; ws += xN * sizeof(bf16_t);
    bf16_t* ab  = (bf16_t*)ws; ws += xN * sizeof(bf16_t);

    // 1) precision conversion
    f32_to_bf16_kernel<<<2048, 256, 0, stream>>>(x,  xb,  xN);
    f32_to_bf16_kernel<<<512,  256, 0, stream>>>(Wq, wqb, wN);
    f32_to_bf16_kernel<<<512,  256, 0, stream>>>(Wk, wkb, wN);
    f32_to_bf16_kernel<<<512,  256, 0, stream>>>(Wv, wvb, wN);
    f32_to_bf16_kernel<<<512,  256, 0, stream>>>(Wo, wob, wN);

    // 2) q,k,v projections (q,k with fused elu+1)
    dim3 gg((unsigned)(M / 256), (unsigned)(D / 64));
    gemm_bf16_kernel<<<gg, 256, 0, stream>>>(xb, wqb, qb, D, D, 1);
    gemm_bf16_kernel<<<gg, 256, 0, stream>>>(xb, wkb, kb, D, D, 1);
    gemm_bf16_kernel<<<gg, 256, 0, stream>>>(xb, wvb, vb, D, D, 0);

    // 3) chunked linear-attention scan: one wave per (b,h)
    linattn_scan_kernel<<<B * 16, 32, 0, stream>>>(qb, kb, vb, ab, S, D);

    // 4) output projection -> f32 d_out
    gemm_bf16_kernel<<<gg, 256, 0, stream>>>(ab, wob, d_out, D, D, 2);
}
